// Transformer_58952721105434
// MI455X (gfx1250) — matrix operand order repaired and checked by reading, not compiled, not run on hardware
//
#include <hip/hip_runtime.h>

// ---------------- Model constants ----------------
#define BSZ 2
#define LSEQ 512
#define DMODEL 1024
#define NHEAD 16
#define DKH 64
#define DFFN 4096
#define VOCAB 32000
#define NSTACK 4
#define NTOK (BSZ * LSEQ)   // 1024 rows

typedef __attribute__((ext_vector_type(16))) __bf16 bf16x16;
typedef __attribute__((ext_vector_type(8)))  float  floatx8;

union AFrag { unsigned u[8]; bf16x16 v; };

// Packed fp32x2 -> bf16x2 (round-to-nearest-even) in one VALU op.
// V_CVT_PK_BF16_F32 exists on CDNA3+ / RDNA4; builtin not exposed, use asm.
__device__ __forceinline__ unsigned pack_bf16x2(float lo, float hi) {
    unsigned r;
    asm("v_cvt_pk_bf16_f32 %0, %1, %2" : "=v"(r) : "v"(lo), "v"(hi));
    return r;
}

// =====================================================================
// Exact-tiling WMMA GEMM. Requires M%128==0, N%TN==0, K%64==0,
// lda/ldb%4==0. No bounds checks anywhere; float4 global staging;
// double-buffered LDS; one barrier per 64-deep K tile; next tile's
// global loads issued between the two K-halves to overlap WMMA.
//   C = act(A @ B(^T) + bias + resid)
// Batch z -> (zb, zh) = (z/nh, z%nh); per-operand offset zb*s?b + zh*s?h.
// =====================================================================
template <bool TRANSB, int TN>
__launch_bounds__(256)
__global__ void gemm_wmma_exact(const float* __restrict__ A,
                                const float* __restrict__ B,
                                const float* __restrict__ bias,
                                const float* __restrict__ resid,
                                float* __restrict__ C,
                                int K, int lda, int ldb, int ldc,
                                long long sAb, long long sAh,
                                long long sBb, long long sBh,
                                long long sCb, long long sCh,
                                int nh, int flags)
{
    constexpr int TM = 128;
    constexpr int TK = 64;
    constexpr int WROWS = (TN == 128) ? 2 : 4;   // waves along M
    constexpr int WCOLS = 8 / WROWS;             // waves along N
    constexpr int MI = (TM / WROWS) / 16;        // M subtiles per wave
    constexpr int NI = (TN / WCOLS) / 16;        // N subtiles per wave
    constexpr int ROWV = TK / 4;                 // float4 per K-row
    constexpr int AVEC = (TM * TK / 4) / 256;    // A float4 per thread
    constexpr int NB4  = TN / 4;
    constexpr int BBLK = ((TK / 2) * NB4) / 256; // 2x4 blocks per thread
    constexpr int BVEC = (TN * TK / 4) / 256;    // B^T float4 per thread
    constexpr int RBN  = TRANSB ? BVEC : 2 * BBLK;

    const bool relu = (flags & 2) != 0;

    const int z  = blockIdx.z;
    const int zb = z / nh, zh = z % nh;
    A += zb * sAb + zh * sAh;
    B += zb * sBb + zh * sBh;
    C += zb * sCb + zh * sCh;
    if (resid) resid += zb * sCb + zh * sCh;

    __shared__ unsigned short As[2][TM][TK];          // 2 x 16KB
    __shared__ unsigned short Bs[2][TK / 2][TN * 2];  // pair-swizzled

    const int tid  = threadIdx.x;
    const int lane = tid & 31;
    const int wave = tid >> 5;
    const int wm   = wave % WROWS;
    const int wn   = wave / WROWS;
    const int half = lane >> 4;
    const int l15  = lane & 15;

    const int blockM = blockIdx.y * TM;
    const int blockN = blockIdx.x * TN;

    const float* Ab = A + (long long)blockM * lda;
    const float* Bb = TRANSB ? (B + (long long)blockN * ldb) : (B + blockN);

    floatx8 acc[MI][NI];
    #pragma unroll
    for (int mi = 0; mi < MI; mi++)
        #pragma unroll
        for (int ni = 0; ni < NI; ni++)
            #pragma unroll
            for (int r = 0; r < 8; r++)
                acc[mi][ni][r] = 0.0f;

    float4 ra[AVEC], rb[RBN];

    auto load_global = [&](int k0) {
        #pragma unroll
        for (int t = 0; t < AVEC; t++) {
            int v  = tid + t * 256;
            int m  = v / ROWV;
            int kq = (v % ROWV) << 2;
            ra[t] = *(const float4*)(Ab + (long long)m * lda + (k0 + kq));
        }
        if (TRANSB) {
            #pragma unroll
            for (int t = 0; t < BVEC; t++) {
                int v  = tid + t * 256;
                int n  = v / ROWV;
                int kq = (v % ROWV) << 2;
                rb[t] = *(const float4*)(Bb + (long long)n * ldb + (k0 + kq));
            }
        } else {
            #pragma unroll
            for (int t = 0; t < BBLK; t++) {
                int bi = tid + t * 256;
                int kb = bi / NB4;
                int nb = (bi % NB4) << 2;
                int k  = k0 + kb * 2;
                rb[2 * t + 0] = *(const float4*)(Bb + (long long)(k + 0) * ldb + nb);
                rb[2 * t + 1] = *(const float4*)(Bb + (long long)(k + 1) * ldb + nb);
            }
        }
    };

    auto store_lds = [&](int p) {
        #pragma unroll
        for (int t = 0; t < AVEC; t++) {
            int v  = tid + t * 256;
            int m  = v / ROWV;
            int kq = (v % ROWV) << 2;
            uint2 w;
            w.x = pack_bf16x2(ra[t].x, ra[t].y);
            w.y = pack_bf16x2(ra[t].z, ra[t].w);
            *(uint2*)&As[p][m][kq] = w;
        }
        if (TRANSB) {
            #pragma unroll
            for (int t = 0; t < BVEC; t++) {
                int v  = tid + t * 256;
                int n  = v / ROWV;
                int kq = (v % ROWV) << 2;
                *(unsigned*)&Bs[p][(kq >> 1) + 0][n * 2] = pack_bf16x2(rb[t].x, rb[t].y);
                *(unsigned*)&Bs[p][(kq >> 1) + 1][n * 2] = pack_bf16x2(rb[t].z, rb[t].w);
            }
        } else {
            #pragma unroll
            for (int t = 0; t < BBLK; t++) {
                int bi = tid + t * 256;
                int kb = bi / NB4;
                int nb = (bi % NB4) << 2;
                uint4 w;
                w.x = pack_bf16x2(rb[2 * t].x, rb[2 * t + 1].x);
                w.y = pack_bf16x2(rb[2 * t].y, rb[2 * t + 1].y);
                w.z = pack_bf16x2(rb[2 * t].z, rb[2 * t + 1].z);
                w.w = pack_bf16x2(rb[2 * t].w, rb[2 * t + 1].w);
                *(uint4*)&Bs[p][kb][nb * 2] = w;
            }
        }
    };

    // frag-read + MI*NI WMMAs for one 32-deep K half at offset ks
    auto compute_half = [&](int p, int ks) {
        AFrag a[MI], b[NI];
        #pragma unroll
        for (int mi = 0; mi < MI; mi++) {
            int m = wm * (TM / WROWS) + mi * 16 + l15;
            #pragma unroll
            for (int j = 0; j < 8; j++) {
                int kk = (j < 4 ? 2 * j : 16 + 2 * (j - 4)) + half * 8;
                a[mi].u[j] = *(const unsigned*)&As[p][m][ks + kk];
            }
        }
        #pragma unroll
        for (int ni = 0; ni < NI; ni++) {
            int n = wn * (TN / WCOLS) + ni * 16 + l15;
            #pragma unroll
            for (int j = 0; j < 8; j++)
                b[ni].u[j] = *(const unsigned*)&Bs[p][(ks >> 1) + (j < 4 ? j : 4 + j) + half * 4][n * 2];
        }
        #pragma unroll
        for (int mi = 0; mi < MI; mi++)
            #pragma unroll
            for (int ni = 0; ni < NI; ni++)
                acc[mi][ni] = __builtin_amdgcn_wmma_f32_16x16x32_bf16(
                    false, a[mi].v, false, b[ni].v,
                    (short)0, acc[mi][ni], false, false);
    };

    const int nk = K / TK;
    load_global(0);

    int p = 0;
    for (int kt = 0; kt < nk; kt++) {
        store_lds(p);
        __syncthreads();

        compute_half(p, 0);

        if (kt + 1 < nk) {
            load_global((kt + 1) * TK);          // overlap with 2nd WMMA half
            if (kt + 2 < nk) {
                int k2 = (kt + 2) * TK;
                __builtin_prefetch(Ab + (long long)(tid / ROWV) * lda + k2, 0, 0);
                __builtin_prefetch(TRANSB ? (Bb + (long long)(tid / ROWV) * ldb + k2)
                                          : (Bb + (long long)k2 * ldb + ((tid % NB4) << 2)),
                                   0, 0);
            }
        }

        compute_half(p, 32);
        p ^= 1;
    }

    // ---- epilogue: lane n = l15, VGPR r -> M = r + half*8 ----
    #pragma unroll
    for (int mi = 0; mi < MI; mi++) {
        #pragma unroll
        for (int ni = 0; ni < NI; ni++) {
            int n = blockN + wn * (TN / WCOLS) + ni * 16 + l15;
            #pragma unroll
            for (int r = 0; r < 8; r++) {
                int m = blockM + wm * (TM / WROWS) + mi * 16 + half * 8 + r;
                float v = acc[mi][ni][r];
                if (bias)  v += bias[n];
                if (resid) v += resid[(long long)m * ldc + n];
                if (relu)  v = fmaxf(v, 0.0f);
                C[(long long)m * ldc + n] = v;
            }
        }
    }
}

// ---------------- Embedding + quirky positional encoding ----------------
__global__ void embed_kernel(const int* __restrict__ tok,
                             const float* __restrict__ emb,
                             float* __restrict__ out, int L)
{
    int row = blockIdx.x;            // b*L + l
    int l   = row % L;
    long long token = tok[row];
    for (int i = threadIdx.x; i < DMODEL; i += blockDim.x) {
        float denum = __expf((2.0f * (float)i / (float)DMODEL) * 9.210340371976184f);
        float x = (float)l / denum;
        float pe = (i & 1) ? __cosf(x) : (1.0f + __sinf(x));
        out[(long long)row * DMODEL + i] = emb[token * DMODEL + i] + pe;
    }
}

// ---------------- scale-fused (causal) softmax over score rows ----------------
// Lk <= 512: each of the 256 threads owns exactly 2 elements (registers).
__global__ void softmax_kernel(float* __restrict__ sc, int Lq, int Lk,
                               float scale, int causal)
{
    int q = blockIdx.x;
    int batch = blockIdx.y;
    float* row = sc + ((long long)batch * Lq + q) * Lk;
    int limit = causal ? (q + 1) : Lk;
    int tid = threadIdx.x;
    __shared__ float red[256];

    int k0 = tid, k1 = tid + 256;
    float x0 = (k0 < limit) ? row[k0] * scale : -3.4e38f;
    float x1 = (k1 < limit) ? row[k1] * scale : -3.4e38f;

    red[tid] = fmaxf(x0, x1); __syncthreads();
    for (int s = 128; s > 0; s >>= 1) {
        if (tid < s) red[tid] = fmaxf(red[tid], red[tid + s]);
        __syncthreads();
    }
    float mx = red[0]; __syncthreads();

    float e0 = (k0 < limit) ? __expf(x0 - mx) : 0.0f;
    float e1 = (k1 < limit) ? __expf(x1 - mx) : 0.0f;
    red[tid] = e0 + e1; __syncthreads();
    for (int s = 128; s > 0; s >>= 1) {
        if (tid < s) red[tid] += red[tid + s];
        __syncthreads();
    }
    float inv = 1.0f / red[0];

    if (k0 < Lk) row[k0] = e0 * inv;
    if (k1 < Lk) row[k1] = e1 * inv;
}

// ---------------- residual-fused LayerNorm (D = 1024, 256 threads) ----------------
__global__ void layernorm_kernel(const float* __restrict__ x,
                                 const float* __restrict__ res,
                                 const float* __restrict__ g,
                                 const float* __restrict__ b,
                                 float* __restrict__ out)
{
    long long row = blockIdx.x;
    const float* xr = x + row * DMODEL;
    const float* rr = res ? res + row * DMODEL : nullptr;
    int tid = threadIdx.x;
    __shared__ float red[256];

    float vals[4];
    float s = 0.0f;
    #pragma unroll
    for (int i = 0; i < 4; i++) {
        int idx = tid + i * 256;
        float v = xr[idx] + (rr ? rr[idx] : 0.0f);
        vals[i] = v; s += v;
    }
    red[tid] = s; __syncthreads();
    for (int st = 128; st > 0; st >>= 1) {
        if (tid < st) red[tid] += red[tid + st];
        __syncthreads();
    }
    float mean = red[0] * (1.0f / DMODEL); __syncthreads();

    float s2 = 0.0f;
    #pragma unroll
    for (int i = 0; i < 4; i++) { float d = vals[i] - mean; s2 += d * d; }
    red[tid] = s2; __syncthreads();
    for (int st = 128; st > 0; st >>= 1) {
        if (tid < st) red[tid] += red[tid + st];
        __syncthreads();
    }
    float inv = rsqrtf(red[0] * (1.0f / DMODEL) + 1e-5f);

    #pragma unroll
    for (int i = 0; i < 4; i++) {
        int idx = tid + i * 256;
        out[row * DMODEL + idx] = g[idx] * (vals[i] - mean) * inv + b[idx];
    }
}

// ---------------- host-side helpers ----------------
static inline void gemm(hipStream_t st, const float* A, const float* B,
                        const float* bias, const float* resid, float* C,
                        int M, int N, int K, int lda, int ldb, int ldc,
                        long long sAb, long long sAh, long long sBb, long long sBh,
                        long long sCb, long long sCh, int nb, int nh, int flags)
{
    // All GEMMs in this model satisfy: M%128==0, K%64==0, lda/ldb%4==0,
    // and N%128==0 (TN=128) or N==64 (TN=64, attn*V only, non-transposed B).
    if (N % 128 == 0) {
        dim3 grid(N / 128, M / 128, nb * nh);
        if (flags & 1)
            gemm_wmma_exact<true, 128><<<grid, 256, 0, st>>>(A, B, bias, resid, C,
                K, lda, ldb, ldc, sAb, sAh, sBb, sBh, sCb, sCh, nh, flags);
        else
            gemm_wmma_exact<false, 128><<<grid, 256, 0, st>>>(A, B, bias, resid, C,
                K, lda, ldb, ldc, sAb, sAh, sBb, sBh, sCb, sCh, nh, flags);
    } else {
        dim3 grid(N / 64, M / 128, nb * nh);
        gemm_wmma_exact<false, 64><<<grid, 256, 0, st>>>(A, B, bias, resid, C,
            K, lda, ldb, ldc, sAb, sAh, sBb, sBh, sCb, sCh, nh, flags);
    }
}

static inline void mha(hipStream_t st, const float* xq, const float* xkv,
                       const float* qkvw, const float* qkvb,
                       const float* ow, const float* ob, int causal,
                       float* q, float* k, float* v, float* sc, float* ctx,
                       float* out)
{
    const long long WSZ = (long long)DMODEL * DMODEL;
    gemm(st, xq,  qkvw + 0 * WSZ, qkvb + 0 * DMODEL, nullptr, q, NTOK, DMODEL,
         DMODEL, DMODEL, DMODEL, DMODEL, 0, 0, 0, 0, 0, 0, 1, 1, 0);
    gemm(st, xkv, qkvw + 1 * WSZ, qkvb + 1 * DMODEL, nullptr, k, NTOK, DMODEL,
         DMODEL, DMODEL, DMODEL, DMODEL, 0, 0, 0, 0, 0, 0, 1, 1, 0);
    gemm(st, xkv, qkvw + 2 * WSZ, qkvb + 2 * DMODEL, nullptr, v, NTOK, DMODEL,
         DMODEL, DMODEL, DMODEL, DMODEL, 0, 0, 0, 0, 0, 0, 1, 1, 0);
    // scores[b,h] = Q[b,:,h*64:+64] @ K[b,:,h*64:+64]^T  -> [32][512][512]
    gemm(st, q, k, nullptr, nullptr, sc, LSEQ, LSEQ, DKH, DMODEL, DMODEL, LSEQ,
         (long long)LSEQ * DMODEL, DKH,
         (long long)LSEQ * DMODEL, DKH,
         (long long)NHEAD * LSEQ * LSEQ, (long long)LSEQ * LSEQ,
         BSZ, NHEAD, /*transB*/ 1);
    softmax_kernel<<<dim3(LSEQ, BSZ * NHEAD), 256, 0, st>>>(sc, LSEQ, LSEQ,
                                                            0.125f, causal);
    // ctx[b,:,h*64:+64] = scores[b,h] @ V[b,:,h*64:+64]   (TN=64 exact path)
    gemm(st, sc, v, nullptr, nullptr, ctx, LSEQ, DKH, LSEQ, LSEQ, DMODEL, DMODEL,
         (long long)NHEAD * LSEQ * LSEQ, (long long)LSEQ * LSEQ,
         (long long)LSEQ * DMODEL, DKH,
         (long long)LSEQ * DMODEL, DKH,
         BSZ, NHEAD, 0);
    gemm(st, ctx, ow, ob, nullptr, out, NTOK, DMODEL, DMODEL, DMODEL, DMODEL,
         DMODEL, 0, 0, 0, 0, 0, 0, 1, 1, 0);
}

extern "C" void kernel_launch(void* const* d_in, const int* in_sizes, int n_in,
                              void* d_out, int out_size, void* d_ws, size_t ws_size,
                              hipStream_t stream)
{
    (void)in_sizes; (void)n_in; (void)out_size; (void)ws_size;
    const int*   s_tok     = (const int*)  d_in[0];
    const int*   t_tok     = (const int*)  d_in[1];
    const float* emb_s_w   = (const float*)d_in[2];
    const float* emb_t_w   = (const float*)d_in[3];
    const float* enc_qkv_w = (const float*)d_in[4];
    const float* enc_qkv_b = (const float*)d_in[5];
    const float* enc_o_w   = (const float*)d_in[6];
    const float* enc_o_b   = (const float*)d_in[7];
    const float* enc_ln_w  = (const float*)d_in[8];
    const float* enc_ln_b  = (const float*)d_in[9];
    const float* enc_w1    = (const float*)d_in[10];
    const float* enc_b1    = (const float*)d_in[11];
    const float* enc_w2    = (const float*)d_in[12];
    const float* enc_b2    = (const float*)d_in[13];
    const float* dec_qkv1_w = (const float*)d_in[14];
    const float* dec_qkv1_b = (const float*)d_in[15];
    const float* dec_o1_w   = (const float*)d_in[16];
    const float* dec_o1_b   = (const float*)d_in[17];
    const float* dec_qkv2_w = (const float*)d_in[18];
    const float* dec_qkv2_b = (const float*)d_in[19];
    const float* dec_o2_w   = (const float*)d_in[20];
    const float* dec_o2_b   = (const float*)d_in[21];
    const float* dec_ln_w   = (const float*)d_in[22];
    const float* dec_ln_b   = (const float*)d_in[23];
    const float* dec_w1     = (const float*)d_in[24];
    const float* dec_b1     = (const float*)d_in[25];
    const float* dec_w2     = (const float*)d_in[26];
    const float* dec_b2     = (const float*)d_in[27];
    const float* out_w      = (const float*)d_in[28];
    const float* out_b      = (const float*)d_in[29];
    float* out = (float*)d_out;

    // ---- workspace carve-up (fp32), ~80 MB total ----
    float* ws = (float*)d_ws;
    float* e   = ws;                        // [1024,1024]
    float* d   = e  + (long long)NTOK * DMODEL;
    float* q   = d  + (long long)NTOK * DMODEL;
    float* k   = q  + (long long)NTOK * DMODEL;
    float* v   = k  + (long long)NTOK * DMODEL;
    float* t1  = v  + (long long)NTOK * DMODEL;   // attention context
    float* t2  = t1 + (long long)NTOK * DMODEL;   // MHA / FFN output
    float* ff  = t2 + (long long)NTOK * DMODEL;   // [1024,4096]
    float* sc  = ff + (long long)NTOK * DFFN;     // [32,512,512] scores

    const long long WSZ  = (long long)DMODEL * DMODEL;
    const long long QKVS = 3 * WSZ;
    const long long F1   = (long long)DMODEL * DFFN;
    const long long F2   = (long long)DFFN * DMODEL;

    embed_kernel<<<NTOK, 256, 0, stream>>>(s_tok, emb_s_w, e, LSEQ);
    embed_kernel<<<NTOK, 256, 0, stream>>>(t_tok, emb_t_w, d, LSEQ);

    for (int i = 0; i < NSTACK; i++) {
        mha(stream, e, e, enc_qkv_w + i * QKVS, enc_qkv_b + i * 3 * DMODEL,
            enc_o_w + i * WSZ, enc_o_b + i * DMODEL, 0, q, k, v, sc, t1, t2);
        layernorm_kernel<<<NTOK, 256, 0, stream>>>(t2, e, enc_ln_w + i * DMODEL,
                                                   enc_ln_b + i * DMODEL, e);
        gemm(stream, e, enc_w1 + i * F1, enc_b1 + i * DFFN, nullptr, ff,
             NTOK, DFFN, DMODEL, DMODEL, DFFN, DFFN, 0,0,0,0,0,0, 1,1, /*relu*/2);
        gemm(stream, ff, enc_w2 + i * F2, enc_b2 + i * DMODEL, nullptr, t2,
             NTOK, DMODEL, DFFN, DFFN, DMODEL, DMODEL, 0,0,0,0,0,0, 1,1, 0);
        layernorm_kernel<<<NTOK, 256, 0, stream>>>(t2, e, enc_ln_w + i * DMODEL,
                                                   enc_ln_b + i * DMODEL, e);
    }

    for (int i = 0; i < NSTACK; i++) {
        mha(stream, d, d, dec_qkv1_w + i * QKVS, dec_qkv1_b + i * 3 * DMODEL,
            dec_o1_w + i * WSZ, dec_o1_b + i * DMODEL, 1, q, k, v, sc, t1, t2);
        layernorm_kernel<<<NTOK, 256, 0, stream>>>(t2, d, dec_ln_w + i * DMODEL,
                                                   dec_ln_b + i * DMODEL, d);
        mha(stream, d, e, dec_qkv2_w + i * QKVS, dec_qkv2_b + i * 3 * DMODEL,
            dec_o2_w + i * WSZ, dec_o2_b + i * DMODEL, 1, q, k, v, sc, t1, t2);
        layernorm_kernel<<<NTOK, 256, 0, stream>>>(t2, d, dec_ln_w + i * DMODEL,
                                                   dec_ln_b + i * DMODEL, d);
        gemm(stream, d, dec_w1 + i * F1, dec_b1 + i * DFFN, nullptr, ff,
             NTOK, DFFN, DMODEL, DMODEL, DFFN, DFFN, 0,0,0,0,0,0, 1,1, 2);
        // d = ffn2(ff) + d2  (residual fused into GEMM epilogue)
        gemm(stream, ff, dec_w2 + i * F2, dec_b2 + i * DMODEL, d, d,
             NTOK, DMODEL, DFFN, DFFN, DMODEL, DMODEL, 0,0,0,0,0,0, 1,1, 0);
    }

    gemm(stream, d, out_w, out_b, nullptr, out, NTOK, VOCAB, DMODEL,
         DMODEL, VOCAB, VOCAB, 0,0,0,0,0,0, 1,1, 0);
}
